// MeshAttentionBlock_51316269252794
// MI455X (gfx1250) — matrix operand order repaired and checked by reading, not compiled, not run on hardware
//
#include <hip/hip_runtime.h>

typedef __attribute__((ext_vector_type(16))) _Float16 v16h;
typedef __attribute__((ext_vector_type(8)))  _Float16 v8h;
typedef __attribute__((ext_vector_type(8)))  float    v8f;
typedef __attribute__((ext_vector_type(4)))  unsigned int u32x4;
typedef __attribute__((ext_vector_type(8)))  int      i32x8;
typedef __attribute__((ext_vector_type(4)))  int      i32x4;

#define NEG_BIG (-3.0e38f)

#if defined(__has_builtin)
#if __has_builtin(__builtin_amdgcn_tensor_load_to_lds) && \
    __has_builtin(__builtin_amdgcn_s_wait_tensorcnt)
#define USE_TDM 1
#endif
#endif
#ifndef USE_TDM
#define USE_TDM 0
#endif

__device__ __forceinline__ v16h cat8(v8h lo, v8h hi) {
    v16h r;
#pragma unroll
    for (int i = 0; i < 8; ++i) { r[i] = lo[i]; r[i + 8] = hi[i]; }
    return r;
}

__device__ __forceinline__ v8f wmma_f16(v16h a, v16h b, v8f c) {
    return __builtin_amdgcn_wmma_f32_16x16x32_f16(false, a, false, b, (short)0, c, false, false);
}

// generic (flat) address of a __shared__ object: low 32 bits are the LDS byte
// offset (flat LDS aperture maps addr[31:0] -> LDS offset).
__device__ __forceinline__ unsigned lds_off_of(const void* p) {
    return (unsigned)(uintptr_t)p;
}

#if USE_TDM
// ---------------------------------------------------------------------------
// TDM descriptor: load a 128-row x 32-half tile (rows strided by `strideElems`
// f16 in global) into LDS with hardware padding: 64B data + 32B pad per row
// -> LDS row stride 96B = 48 halves (matches GLDK below).
// D# group0: count=1|type=2, lds_addr, global_addr.  D# group1: data_size=2B,
// pad_enable, pad_interval=16 DW (64B), pad_amount=8 DW (32B), tensor dims,
// tile 32x128, dim0 stride.  This toolchain's builtin takes 6 args
// (g0, g1, g2, g3, extra group, cpol); groups 2/3/extra unused for 2-D tiles.
// ---------------------------------------------------------------------------
__device__ __forceinline__ void tdm_load_tile_128x32(
    const _Float16* gsrc, unsigned lds_off, unsigned strideElems) {
    const unsigned long long ga = (unsigned long long)(uintptr_t)gsrc;
    u32x4 g0;
    g0[0] = 1u;                                    // count=1 (valid), user mode
    g0[1] = lds_off;                               // LDS byte address
    g0[2] = (unsigned)(ga & 0xFFFFFFFFu);          // global addr [31:0]
    g0[3] = (unsigned)((ga >> 32) & 0x01FFFFFFu) | (2u << 30);  // [56:32]|type=2
    i32x8 g1;
    g1[0] = (int)((1u << 16)      // data_size = 2 bytes
                | (1u << 20)      // pad_enable
                | (3u << 22)      // pad_interval: 16 DWORDs = 64B
                | (7u << 25));    // pad_amount:    8 DWORDs = 32B
    g1[1] = (int)((strideElems & 0xFFFFu) << 16);  // tensor_dim0 lo16
    g1[2] = (int)((strideElems >> 16) & 0xFFFFu);  // tensor_dim0 hi16, dim1 lo16=0
    g1[3] = (int)(16u | (32u << 16));              // tensor_dim1=1<<20, tile_dim0=32
    g1[4] = (int)(128u);                           // tile_dim1=128, tile_dim2=0
    g1[5] = (int)strideElems;                      // tensor_dim0_stride lo32
    g1[6] = 0;
    g1[7] = 0;
    const i32x4 z4 = {0, 0, 0, 0};
    const i32x8 z8 = {0, 0, 0, 0, 0, 0, 0, 0};
    __builtin_amdgcn_tensor_load_to_lds(g0, g1, z4, z4, z8, 0);
}
#endif

// ---------------------------------------------------------------------------
// 1) adaLN parameter GEMM: p[sel][b][0:2048] = silu(cond[b]) @ ada_w + ada_b
// ---------------------------------------------------------------------------
__global__ __launch_bounds__(256) void ada_params_kernel(
    const float* __restrict__ cond,
    const float* __restrict__ w1, const float* __restrict__ b1,
    const float* __restrict__ w2, const float* __restrict__ b2,
    float* __restrict__ p) {
    __shared__ float sc[512];
    const int blk   = blockIdx.x;        // 64 blocks: sel(2) x b(4) x chunk(8)
    const int chunk = blk & 7;
    const int b     = (blk >> 3) & 3;
    const int sel   = blk >> 5;
    const float* w  = sel ? w2 : w1;
    const float* bb = sel ? b2 : b1;
    for (int i = threadIdx.x; i < 512; i += 256) {
        float c = cond[b * 512 + i];
        sc[i] = c / (1.0f + expf(-c));
    }
    __syncthreads();
    const int col = chunk * 256 + threadIdx.x;
    float acc = bb[col];
    for (int k = 0; k < 512; ++k) acc = fmaf(sc[k], w[(size_t)k * 2048 + col], acc);
    p[((size_t)sel * 4 + b) * 2048 + col] = acc;
}

// ---------------------------------------------------------------------------
// 2) adaLN: per-token layernorm + scale/shift; emits f16 activations.
// ---------------------------------------------------------------------------
__global__ __launch_bounds__(256) void adaln_kernel(
    const float* __restrict__ x, const float* __restrict__ p,
    _Float16* __restrict__ out) {
    __shared__ float red[256];
    const int row = blockIdx.x;          // 0..4095
    const int b   = row >> 10;
    const int t   = threadIdx.x;
    const float* xr = x + (size_t)row * 1024;
    float v[4];
    float s = 0.0f;
#pragma unroll
    for (int i = 0; i < 4; ++i) { v[i] = xr[t + i * 256]; s += v[i]; }
    red[t] = s; __syncthreads();
    for (int o = 128; o > 0; o >>= 1) { if (t < o) red[t] += red[t + o]; __syncthreads(); }
    const float mu = red[0] * (1.0f / 1024.0f);
    __syncthreads();
    s = 0.0f;
#pragma unroll
    for (int i = 0; i < 4; ++i) { float d = v[i] - mu; s += d * d; }
    red[t] = s; __syncthreads();
    for (int o = 128; o > 0; o >>= 1) { if (t < o) red[t] += red[t + o]; __syncthreads(); }
    const float inv = rsqrtf(red[0] * (1.0f / 1024.0f) + 1e-5f);
    const float* scp = p + (size_t)b * 2048;
    const float* shp = scp + 1024;
#pragma unroll
    for (int i = 0; i < 4; ++i) {
        const int c = t + i * 256;
        out[(size_t)row * 1024 + c] =
            (_Float16)((v[i] - mu) * inv * (1.0f + scp[c]) + shp[c]);
    }
}

// ---------------------------------------------------------------------------
// 3) weight convert+transpose: fp32 [K,N] row-major -> f16 [N,K] row-major
// ---------------------------------------------------------------------------
__global__ __launch_bounds__(256) void convert_wT_kernel(
    const float* __restrict__ w, _Float16* __restrict__ o, int K, int N) {
    const size_t i = (size_t)blockIdx.x * 256 + threadIdx.x;  // over N*K
    const int n = (int)(i / (size_t)K);
    const int k = (int)(i % (size_t)K);
    o[i] = (_Float16)w[(size_t)k * N + n];
}

// ---------------------------------------------------------------------------
// 4) WMMA GEMM: C[M,N] = A_f16[M,K] * Bt_f16[N,K]^T, fp32 accumulate.
//    256 threads = 8 waves, 128x128 tile, K-step 32.
//    Double-buffered LDS staging via TDM (tensor_load_to_lds + TENSORcnt),
//    cooperative-load fallback if the builtin is unavailable.
//    EPI: 0 = store f16; 1 = +bias +residual fp32; 2 = +bias GELU f16
// ---------------------------------------------------------------------------
#define GLDK 48
template <int EPI>
__global__ __launch_bounds__(256) void gemm_f16_kernel(
    const _Float16* __restrict__ A, const _Float16* __restrict__ Bt,
    void* __restrict__ Cout, const float* __restrict__ bias,
    const float* __restrict__ res, int M, int N, int K) {
    __shared__ alignas(64) _Float16 As[2][128 * GLDK];
    __shared__ alignas(64) _Float16 Bs[2][128 * GLDK];
    const int tilesN = N >> 7;
    const int bm = blockIdx.x / tilesN;
    const int bn = blockIdx.x % tilesN;
    const int t = threadIdx.x;
    const int wave = t >> 5, lane = t & 31;
    const int lhalf = lane >> 4, lmod = lane & 15;

    v8f acc[8];
#pragma unroll
    for (int n = 0; n < 8; ++n)
#pragma unroll
        for (int r = 0; r < 8; ++r) acc[n][r] = 0.0f;

    const _Float16* Ag = A  + (size_t)(bm * 128) * K;
    const _Float16* Bg = Bt + (size_t)(bn * 128) * K;
    const int steps = K >> 5;

#if USE_TDM
    if (wave == 0) {
        tdm_load_tile_128x32(Ag, lds_off_of(&As[0][0]), (unsigned)K);
        tdm_load_tile_128x32(Bg, lds_off_of(&Bs[0][0]), (unsigned)K);
    }
#else
    const int srow = t >> 1;            // 0..127
    const int sseg = (t & 1) * 16;      // 0 or 16
    {
        const _Float16* ag = Ag + (size_t)srow * K + sseg;
        const _Float16* bg = Bg + (size_t)srow * K + sseg;
        *(v8h*)&As[0][srow * GLDK + sseg]     = *(const v8h*)ag;
        *(v8h*)&As[0][srow * GLDK + sseg + 8] = *(const v8h*)(ag + 8);
        *(v8h*)&Bs[0][srow * GLDK + sseg]     = *(const v8h*)bg;
        *(v8h*)&Bs[0][srow * GLDK + sseg + 8] = *(const v8h*)(bg + 8);
    }
#endif

    for (int i = 0; i < steps; ++i) {
        const int cur = i & 1;
#if USE_TDM
        if (wave == 0) __builtin_amdgcn_s_wait_tensorcnt(0);
#endif
        __syncthreads();   // buf[cur] ready; buf[cur^1] fully consumed
#if USE_TDM
        if (wave == 0 && (i + 1) < steps) {
            tdm_load_tile_128x32(Ag + (i + 1) * 32,
                                 lds_off_of(&As[cur ^ 1][0]), (unsigned)K);
            tdm_load_tile_128x32(Bg + (i + 1) * 32,
                                 lds_off_of(&Bs[cur ^ 1][0]), (unsigned)K);
        }
#else
        if ((i + 1) < steps) {
            const int k0 = (i + 1) * 32;
            const _Float16* ag = Ag + (size_t)srow * K + k0 + sseg;
            const _Float16* bg = Bg + (size_t)srow * K + k0 + sseg;
            __builtin_prefetch(ag + 32, 0, 0);
            __builtin_prefetch(bg + 32, 0, 0);
            *(v8h*)&As[cur ^ 1][srow * GLDK + sseg]     = *(const v8h*)ag;
            *(v8h*)&As[cur ^ 1][srow * GLDK + sseg + 8] = *(const v8h*)(ag + 8);
            *(v8h*)&Bs[cur ^ 1][srow * GLDK + sseg]     = *(const v8h*)bg;
            *(v8h*)&Bs[cur ^ 1][srow * GLDK + sseg + 8] = *(const v8h*)(bg + 8);
        }
#endif
        const _Float16* Asb = &As[cur][0];
        const _Float16* Bsb = &Bs[cur][0];

        // A fragment: lane = row M, 16 contiguous K halves selected by lane/16
        v16h af;
        {
            const int row = wave * 16 + lmod;
            v8h a0 = *(const v8h*)&Asb[row * GLDK + lhalf * 16];
            v8h a1 = *(const v8h*)&Asb[row * GLDK + lhalf * 16 + 8];
            af = cat8(a0, a1);
        }
#pragma unroll
        for (int n = 0; n < 8; ++n) {
            const int row = n * 16 + lmod;   // Bt row = output column
            const v16h bfrag = *(const v16h*)&Bsb[row * GLDK + lhalf * 16];
            acc[n] = wmma_f16(af, bfrag, acc[n]);
        }
        __syncthreads();   // done reading buf[cur] before it is refilled
    }

    // epilogue: C layout  M=(lane/16)*8+r, N=lane%16
#pragma unroll
    for (int n = 0; n < 8; ++n) {
        const int col = bn * 128 + n * 16 + lmod;
#pragma unroll
        for (int r = 0; r < 8; ++r) {
            const int row = bm * 128 + wave * 16 + lhalf * 8 + r;
            float v = acc[n][r];
            if (EPI == 0) {
                ((_Float16*)Cout)[(size_t)row * N + col] = (_Float16)v;
            } else if (EPI == 1) {
                v += bias[col] + res[(size_t)row * N + col];
                ((float*)Cout)[(size_t)row * N + col] = v;
            } else {  // bias + exact GELU -> f16
                v += bias[col];
                v = 0.5f * v * (1.0f + erff(v * 0.70710678118f));
                ((_Float16*)Cout)[(size_t)row * N + col] = (_Float16)v;
            }
        }
    }
}

// ---------------------------------------------------------------------------
// 5) Flash attention per (b, h, 64-query tile). 128 threads = 4 waves.
//    S = Q K^T via WMMA (contract d), fused scale + edge bias + mask,
//    online softmax, P through LDS (C->A layout), P*V via WMMA with V
//    stored transposed in LDS. Output f16 [B*V, 1024].
// ---------------------------------------------------------------------------
__global__ __launch_bounds__(128) void flash_attn_kernel(
    const _Float16* __restrict__ qkv, const int* __restrict__ edge_index,
    const unsigned char* __restrict__ amask, const float* __restrict__ edge_table,
    _Float16* __restrict__ aout) {
    constexpr int Vn = 1024, Hh = 16;
    __shared__ alignas(64) _Float16 Ks[32][64];     // [j][d]
    __shared__ alignas(64) _Float16 VsT[64][32];    // [d][j]
    __shared__ alignas(64) _Float16 Ps[4][16][32];  // per-wave P tile [q][j]
    __shared__ float etab[64];

    const int blk = blockIdx.x;
    const int qt  = blk & 15;
    const int h   = (blk >> 4) & 15;
    const int b   = blk >> 8;
    const int t = threadIdx.x;
    const int wave = t >> 5, lane = t & 31;
    const int lhalf = lane >> 4, lmod = lane & 15;
    if (t < 64) etab[t] = edge_table[t];

    const size_t tokStride = 3072;
    const _Float16* qbase = qkv + (size_t)b * Vn * tokStride + h * 64;
    const _Float16* kbase = qbase + 1024;
    const _Float16* vbase = qbase + 2048;
    const int q0 = qt * 64;

    // Q fragments for this wave's 16 rows (registers, reused across all j)
    v16h Qf[2];
    {
        const int qrow = q0 + wave * 16 + lmod;
#pragma unroll
        for (int dstep = 0; dstep < 2; ++dstep) {
            const _Float16* qp = qbase + (size_t)qrow * tokStride + dstep * 32;
            v8h a0 = *(const v8h*)(qp + lhalf * 16);
            v8h a1 = *(const v8h*)(qp + lhalf * 16 + 8);
            Qf[dstep] = cat8(a0, a1);
        }
    }

    float mrow[8], lrow[8];
    v8f Oa[4];
#pragma unroll
    for (int r = 0; r < 8; ++r) { mrow[r] = NEG_BIG; lrow[r] = 0.0f; }
#pragma unroll
    for (int d = 0; d < 4; ++d)
#pragma unroll
        for (int r = 0; r < 8; ++r) Oa[d][r] = 0.0f;

    const float scale = 0.125f;  // 64^-0.5
    const int srow = t >> 2;             // 0..31 key row
    const int sseg = (t & 3) * 16;       // 0,16,32,48

    for (int j0 = 0; j0 < Vn; j0 += 32) {
        __syncthreads();
        {   // stage K rows (row-major) and V rows (transposed scatter)
            const _Float16* kp = kbase + (size_t)(j0 + srow) * tokStride + sseg;
            *(v8h*)&Ks[srow][sseg]     = *(const v8h*)kp;
            *(v8h*)&Ks[srow][sseg + 8] = *(const v8h*)(kp + 8);
            const _Float16* vp = vbase + (size_t)(j0 + srow) * tokStride + sseg;
            v8h v0 = *(const v8h*)vp;
            v8h v1 = *(const v8h*)(vp + 8);
#pragma unroll
            for (int i = 0; i < 8; ++i) {
                VsT[sseg + i][srow]     = v0[i];
                VsT[sseg + 8 + i][srow] = v1[i];
            }
        }
        __syncthreads();

        // S = Q K^T for two 16-column tiles; fuse scale + edge bias + mask
        v8f S[2];
#pragma unroll
        for (int jb = 0; jb < 2; ++jb) {
            v8f s;
#pragma unroll
            for (int r = 0; r < 8; ++r) s[r] = 0.0f;
#pragma unroll
            for (int dstep = 0; dstep < 2; ++dstep) {
                const int row = jb * 16 + lmod;
                const v16h kf = *(const v16h*)&Ks[row][dstep * 32 + lhalf * 16];
                s = wmma_f16(Qf[dstep], kf, s);
            }
            const int j = j0 + jb * 16 + lmod;
            const unsigned char mk = amask[b * Vn + j];
#pragma unroll
            for (int r = 0; r < 8; ++r) {
                const int q = q0 + wave * 16 + lhalf * 8 + r;
                const int ei = edge_index[((size_t)(b * Vn + q)) * Vn + j];
                float v = s[r] * scale + etab[ei * Hh + h];
                s[r] = mk ? v : NEG_BIG;
            }
            S[jb] = s;
        }

        // online softmax per row (rows split by lane-half; reduce over 16 lanes)
#pragma unroll
        for (int r = 0; r < 8; ++r) {
            float mx = fmaxf(S[0][r], S[1][r]);
#pragma unroll
            for (int m = 1; m < 16; m <<= 1) mx = fmaxf(mx, __shfl_xor(mx, m, 32));
            const float mnew  = fmaxf(mrow[r], mx);
            const float alpha = expf(mrow[r] - mnew);
            const float p0 = expf(S[0][r] - mnew);
            const float p1 = expf(S[1][r] - mnew);
            S[0][r] = p0; S[1][r] = p1;
            float rs = p0 + p1;
#pragma unroll
            for (int m = 1; m < 16; m <<= 1) rs += __shfl_xor(rs, m, 32);
            lrow[r] = lrow[r] * alpha + rs;
            mrow[r] = mnew;
#pragma unroll
            for (int d = 0; d < 4; ++d) Oa[d][r] *= alpha;
        }

        // P: C layout -> LDS -> A layout (per-wave region, same-wave DS order)
#pragma unroll
        for (int jb = 0; jb < 2; ++jb)
#pragma unroll
            for (int r = 0; r < 8; ++r)
                Ps[wave][lhalf * 8 + r][jb * 16 + lmod] = (_Float16)S[jb][r];

        v16h Pf;
        {
            v8h p0 = *(const v8h*)&Ps[wave][lmod][lhalf * 16];
            v8h p1 = *(const v8h*)&Ps[wave][lmod][lhalf * 16 + 8];
            Pf = cat8(p0, p1);
        }
        // O += P * V  (contract j=32; V^T fragments contiguous in LDS)
#pragma unroll
        for (int db = 0; db < 4; ++db) {
            const int drow = db * 16 + lmod;
            const v16h vf = *(const v16h*)&VsT[drow][lhalf * 16];
            Oa[db] = wmma_f16(Pf, vf, Oa[db]);
        }
    }

    // normalize and store f16 [B*V, 1024]
#pragma unroll
    for (int db = 0; db < 4; ++db)
#pragma unroll
        for (int r = 0; r < 8; ++r) {
            const int q = q0 + wave * 16 + lhalf * 8 + r;
            const int col = h * 64 + db * 16 + lmod;
            aout[(size_t)(b * Vn + q) * 1024 + col] =
                (_Float16)(Oa[db][r] / lrow[r]);
        }
}

// ---------------------------------------------------------------------------
// host-side orchestration
// ---------------------------------------------------------------------------
extern "C" void kernel_launch(void* const* d_in, const int* in_sizes, int n_in,
                              void* d_out, int out_size, void* d_ws, size_t ws_size,
                              hipStream_t stream) {
    (void)in_sizes; (void)n_in; (void)out_size; (void)ws_size;
    const float* x        = (const float*)d_in[0];
    const float* cond     = (const float*)d_in[1];
    const int*   eidx     = (const int*)d_in[2];
    const unsigned char* amask = (const unsigned char*)d_in[3];
    const float* ada1_w   = (const float*)d_in[4];
    const float* ada1_b   = (const float*)d_in[5];
    const float* ada2_w   = (const float*)d_in[6];
    const float* ada2_b   = (const float*)d_in[7];
    const float* w_qkv    = (const float*)d_in[8];
    const float* w_proj   = (const float*)d_in[9];
    const float* b_proj   = (const float*)d_in[10];
    const float* edge_tab = (const float*)d_in[11];
    const float* mlp_w1   = (const float*)d_in[12];
    const float* mlp_b1   = (const float*)d_in[13];
    const float* mlp_w2   = (const float*)d_in[14];
    const float* mlp_b2   = (const float*)d_in[15];

    char* ws = (char*)d_ws;
    const size_t OFF_P    = 0;                    // 64 KB  ada params p[2][4][2048]
    const size_t OFF_H    = 65536;                // 8 MB   h1 then h2 (f16)
    const size_t OFF_W    = OFF_H    + 8388608;   // 8 MB   current f16 transposed weight
    const size_t OFF_QKV  = OFF_W    + 8388608;   // 32 MB  qkv then mlp-mid (f16)
    const size_t OFF_AOUT = OFF_QKV  + 33554432;  // 8 MB   attention out (f16)
    const size_t OFF_X1   = OFF_AOUT + 8388608;   // 16 MB  residual x1 (fp32)

    float*     p    = (float*)(ws + OFF_P);
    _Float16*  hbuf = (_Float16*)(ws + OFF_H);
    _Float16*  wT   = (_Float16*)(ws + OFF_W);
    _Float16*  qkv  = (_Float16*)(ws + OFF_QKV);
    _Float16*  mid  = qkv;                        // reuse after attention
    _Float16*  aout = (_Float16*)(ws + OFF_AOUT);
    float*     x1   = (float*)(ws + OFF_X1);
    float*     out  = (float*)d_out;

    const int M = 4096;  // B*V

    ada_params_kernel<<<64, 256, 0, stream>>>(cond, ada1_w, ada1_b, ada2_w, ada2_b, p);
    adaln_kernel<<<M, 256, 0, stream>>>(x, p, hbuf);
    convert_wT_kernel<<<(3072 * 1024) / 256, 256, 0, stream>>>(w_qkv, wT, 1024, 3072);
    gemm_f16_kernel<0><<<(M / 128) * (3072 / 128), 256, 0, stream>>>(
        hbuf, wT, qkv, nullptr, nullptr, M, 3072, 1024);
    flash_attn_kernel<<<1024, 128, 0, stream>>>(qkv, eidx, amask, edge_tab, aout);
    convert_wT_kernel<<<(1024 * 1024) / 256, 256, 0, stream>>>(w_proj, wT, 1024, 1024);
    gemm_f16_kernel<1><<<(M / 128) * (1024 / 128), 256, 0, stream>>>(
        aout, wT, x1, b_proj, x, M, 1024, 1024);
    adaln_kernel<<<M, 256, 0, stream>>>(x1, p + 4 * 2048, hbuf);
    convert_wT_kernel<<<(4096 * 1024) / 256, 256, 0, stream>>>(mlp_w1, wT, 1024, 4096);
    gemm_f16_kernel<2><<<(M / 128) * (4096 / 128), 256, 0, stream>>>(
        hbuf, wT, mid, mlp_b1, nullptr, M, 4096, 1024);
    convert_wT_kernel<<<(1024 * 4096) / 256, 256, 0, stream>>>(mlp_w2, wT, 4096, 1024);
    gemm_f16_kernel<1><<<(M / 128) * (1024 / 128), 256, 0, stream>>>(
        mid, wT, out, mlp_b2, x1, M, 1024, 4096);
}
